// PointnetSAModuleMSG_WithSampling_54735063220283
// MI455X (gfx1250) — compile-verified, matrix operand order checked by reading, not run on hardware
//
#include <hip/hip_runtime.h>
#include <hip/hip_bf16.h>
#include <hip/hip_fp16.h>

// PointNet++ SA-MSG module for MI455X (gfx1250, wave32, WMMA).
// All matmuls run through v_wmma_f32_16x16x32_f16 (f16 A/B, f32 accum).
// Weights are pre-swizzled into WMMA B-fragment layout; each wave computes a
// 16x64 output strip (4 WMMA tiles) to amortize A loads and address math.

#define BB   8
#define NN   8192
#define SS   2048
#define CF   64
#define NS0  16
#define NS1  32
#define RAD0 0.4f
#define RAD1 0.8f
#define EPSB 1e-5f

typedef __attribute__((ext_vector_type(16))) _Float16 v16h;
typedef __attribute__((ext_vector_type(8)))  _Float16 v8h;
typedef __attribute__((ext_vector_type(8)))  float    v8f;

// ---------------- FPS: one block per batch, 2048 serial argmax steps --------
__global__ void fps_kernel(const float* __restrict__ xyz, float* __restrict__ dist,
                           float* __restrict__ new_xyz) {
  const int b = blockIdx.x;
  const int t = threadIdx.x;
  const float* X = xyz + (size_t)b * NN * 3;
  float* D = dist + (size_t)b * NN;
  __shared__ float sval[256];
  __shared__ int   sidx[256];
  for (int n = t; n < NN; n += 256) D[n] = 1e10f;
  __syncthreads();
  int last = 0;
  for (int it = 0; it < SS; ++it) {
    if (t == 0) {
      new_xyz[((size_t)b * SS + it) * 3 + 0] = X[last * 3 + 0];
      new_xyz[((size_t)b * SS + it) * 3 + 1] = X[last * 3 + 1];
      new_xyz[((size_t)b * SS + it) * 3 + 2] = X[last * 3 + 2];
    }
    const float lx = X[last * 3 + 0], ly = X[last * 3 + 1], lz = X[last * 3 + 2];
    float best = -1.0f; int bi = 0;
    for (int n = t; n < NN; n += 256) {
      float dx = X[n * 3 + 0] - lx, dy = X[n * 3 + 1] - ly, dz = X[n * 3 + 2] - lz;
      float dn = fminf(D[n], dx * dx + dy * dy + dz * dz);
      D[n] = dn;
      if (dn > best) { best = dn; bi = n; }
    }
    sval[t] = best; sidx[t] = bi;
    __syncthreads();
    for (int off = 128; off > 0; off >>= 1) {
      if (t < off) {
        float ov = sval[t + off]; int oi = sidx[t + off];
        if (ov > sval[t] || (ov == sval[t] && oi < sidx[t])) { sval[t] = ov; sidx[t] = oi; }
      }
      __syncthreads();
    }
    last = sidx[0];
    __syncthreads();
  }
}

// ------------- ball query + gather: one wave32 per center -------------------
__global__ void ballq_gather_kernel(const float* __restrict__ xyz,
                                    const float* __restrict__ feats,
                                    const float* __restrict__ new_xyz,
                                    _Float16* __restrict__ act,
                                    int ns, float r2) {
  const int w    = threadIdx.x >> 5;
  const int lane = threadIdx.x & 31;
  const int cid  = blockIdx.x * 8 + w;
  const int b    = cid / SS;
  __shared__ int grp[8][32];
  const float* X = xyz   + (size_t)b * NN * 3;
  const float* F = feats + (size_t)b * NN * CF;
  const float cx = new_xyz[(size_t)cid * 3 + 0];
  const float cy = new_xyz[(size_t)cid * 3 + 1];
  const float cz = new_xyz[(size_t)cid * 3 + 2];
  int cnt = 0;
  for (int base = 0; base < NN && cnt < ns; base += 32) {
    const int n = base + lane;
    float dx = X[n * 3 + 0] - cx, dy = X[n * 3 + 1] - cy, dz = X[n * 3 + 2] - cz;
    bool in = (dx * dx + dy * dy + dz * dz) < r2;
    unsigned mask = (unsigned)__ballot(in);
    int pos = cnt + __popc(mask & ((1u << lane) - 1u));
    if (in && pos < ns) grp[w][pos] = n;
    cnt += __popc(mask);
  }
  __syncthreads();
  int idx0 = (cnt > 0) ? grp[w][0] : 0;
  if (lane < ns && lane >= cnt) grp[w][lane] = idx0;
  __syncthreads();
  const size_t rowbase = (size_t)cid * ns;
  for (int slot = 0; slot < ns; ++slot) {
    const int p = grp[w][slot];
    _Float16* dst = act + (rowbase + slot) * 96;
    for (int c = lane; c < 96; c += 32) {
      float v;
      if (c < 3)       v = X[p * 3 + c] - (c == 0 ? cx : (c == 1 ? cy : cz));
      else if (c < 67) v = F[(size_t)p * CF + (c - 3)];
      else             v = 0.0f;
      dst[c] = (_Float16)v;
    }
  }
}

// ------- weight f32 -> f16 in WMMA B-fragment layout, K padded to mult 32 ---
// Fragment layout: frag[((nt*KS + ks)*32 + lane)*16 + i] where
//   n = nt*16 + (lane&15), k = ks*32 + (lane>>4)*8 + (i<8 ? i : i+8)
// so the GEMM reads one contiguous 32B v16h per lane per k-step.
__global__ void convert_w_kernel(const float* __restrict__ w, _Float16* __restrict__ o,
                                 int K, int Kpad, int Ncols) {
  const int tid = blockIdx.x * blockDim.x + threadIdx.x;
  if (tid >= Kpad * Ncols) return;
  const int KS = Kpad >> 5;
  const int i    = tid & 15;
  const int lane = (tid >> 4) & 31;
  const int rest = tid >> 9;
  const int ks = rest % KS;
  const int nt = rest / KS;
  const int n = nt * 16 + (lane & 15);
  const int k = ks * 32 + (lane >> 4) * 8 + (i < 8 ? i : i + 8);
  o[tid] = (k < K) ? (_Float16)w[(size_t)k * Ncols + n] : (_Float16)0.0f;
}

// ------------- WMMA GEMM: one wave per 16(M)x64(N) output strip -------------
// A: [rows, lda] f16 row-major.  Wfrag: B-fragment layout (see above).
// Out: [rows, ldo] f16.  rows%16==0, K%32==0, Ncols%64==0.
__global__ void gemm_wmma_kernel(const _Float16* __restrict__ A, int lda,
                                 const _Float16* __restrict__ Wfrag,
                                 _Float16* __restrict__ Out, int ldo,
                                 const float* __restrict__ bias,
                                 int rows, int K, int Ncols) {
  const int wid  = (blockIdx.x * blockDim.x + threadIdx.x) >> 5;
  const int lane = threadIdx.x & 31;
  const int Mtiles = rows >> 4;
  const int NtGroups = Ncols >> 6;           // 4 N-tiles per wave
  if (wid >= Mtiles * NtGroups) return;
  const int ng = wid % NtGroups;
  const int mt = wid / NtGroups;
  const int nt0 = ng * 4;
  const int KS = K >> 5;
  const int hl = lane & 15;
  const int hi = lane >> 4;
  const int arow = mt * 16 + hl;
  v8f acc0 = {}, acc1 = {}, acc2 = {}, acc3 = {};
  const _Float16* arp = A + (size_t)arow * lda + hi * 8;
  for (int ks = 0; ks < KS; ++ks) {
    // A fragment: lane holds K = kb..kb+7 and kb+16..kb+23 (ISA 16-bit A layout)
    const _Float16* ap = arp + ks * 32;
    v8h a0 = *(const v8h*)(ap);
    v8h a1 = *(const v8h*)(ap + 16);
    v16h a;
#pragma unroll
    for (int i = 0; i < 8; ++i) { a[i] = a0[i]; a[i + 8] = a1[i]; }
    // 4 pre-swizzled B fragments: one contiguous 32B load each
    const _Float16* wp = Wfrag + ((size_t)(nt0 * KS + ks) * 32 + lane) * 16;
    const size_t fstride = (size_t)KS * 32 * 16;
    v16h b0 = *(const v16h*)(wp);
    v16h b1 = *(const v16h*)(wp + fstride);
    v16h b2 = *(const v16h*)(wp + 2 * fstride);
    v16h b3 = *(const v16h*)(wp + 3 * fstride);
    acc0 = __builtin_amdgcn_wmma_f32_16x16x32_f16(false, a, false, b0, (short)0, acc0, false, false);
    acc1 = __builtin_amdgcn_wmma_f32_16x16x32_f16(false, a, false, b1, (short)0, acc1, false, false);
    acc2 = __builtin_amdgcn_wmma_f32_16x16x32_f16(false, a, false, b2, (short)0, acc2, false, false);
    acc3 = __builtin_amdgcn_wmma_f32_16x16x32_f16(false, a, false, b3, (short)0, acc3, false, false);
  }
  // C/D layout: VGPR r holds M = r + 8*(lane>=16), N = nt*16 + (lane&15)
#pragma unroll
  for (int j = 0; j < 4; ++j) {
    const v8f* accs[4] = {&acc0, &acc1, &acc2, &acc3};
    const v8f av = *accs[j];
    const int ncol = (nt0 + j) * 16 + hl;
    const float bv = bias ? bias[ncol] : 0.0f;
#pragma unroll
    for (int r = 0; r < 8; ++r) {
      const int orow = mt * 16 + hi * 8 + r;
      Out[(size_t)orow * ldo + ncol] = (_Float16)(av[r] + bv);
    }
  }
}

// ------------- per-channel sum / sumsq for BatchNorm ------------------------
__global__ void stats_kernel(const _Float16* __restrict__ Z, int ldz,
                             int rows, int Ncols, float* __restrict__ stats) {
  const int total = gridDim.x * blockDim.x;     // multiple of 256 >= Ncols
  const int tid = blockIdx.x * blockDim.x + threadIdx.x;
  const int n = tid % Ncols;
  const int rstep = total / Ncols;
  float s = 0.0f, ss = 0.0f;
  for (int r = tid / Ncols; r < rows; r += rstep) {
    float v = (float)Z[(size_t)r * ldz + n];
    s += v; ss += v * v;
  }
  atomicAdd(&stats[n], s);
  atomicAdd(&stats[Ncols + n], ss);
}

// ------------- BN (training stats) + ReLU; dual f16/f32 outputs -------------
__global__ void bn_relu_kernel(const _Float16* __restrict__ Z, int ldz,
                               _Float16* __restrict__ o16, int ld16,
                               float* __restrict__ o32, int ld32,
                               const float* __restrict__ stats,
                               size_t rows, int Ncols, float invCount) {
  const size_t stride = (size_t)gridDim.x * blockDim.x;
  const size_t total = rows * (size_t)Ncols;
  for (size_t tid = blockIdx.x * (size_t)blockDim.x + threadIdx.x;
       tid < total; tid += stride) {
    const int n = (int)(tid % Ncols);
    const size_t r = tid / Ncols;
    const float mean = stats[n] * invCount;
    const float var  = stats[Ncols + n] * invCount - mean * mean;
    float v = ((float)Z[r * ldz + n] - mean) * rsqrtf(var + EPSB);
    v = fmaxf(v, 0.0f);
    if (o16) o16[r * ld16 + n] = (_Float16)v;
    if (o32) o32[r * ld32 + n] = v;
  }
}

// ------------- max over group -> concat buffer ------------------------------
__global__ void maxpool_kernel(const _Float16* __restrict__ actv, int ns,
                               _Float16* __restrict__ concat, int coloff) {
  const int tid = blockIdx.x * blockDim.x + threadIdx.x;
  if (tid >= BB * SS * 128) return;
  const int s = tid / 128, ch = tid % 128;
  float m = (float)actv[((size_t)s * ns) * 128 + ch];
  for (int j = 1; j < ns; ++j)
    m = fmaxf(m, (float)actv[((size_t)s * ns + j) * 128 + ch]);
  concat[(size_t)s * 256 + coloff + ch] = (_Float16)m;
}

// ------------- tiny classification head (256 -> 3), f32 ---------------------
__global__ void cls_kernel(const float* __restrict__ c, const float* __restrict__ wc,
                           const float* __restrict__ bc, float* __restrict__ out) {
  const int tid = blockIdx.x * blockDim.x + threadIdx.x;
  if (tid >= BB * SS * 3) return;
  const int row = tid / 3, j = tid % 3;
  float acc = bc[j];
  for (int k = 0; k < 256; ++k) acc += c[(size_t)row * 256 + k] * wc[k * 3 + j];
  out[tid] = acc;
}

extern "C" void kernel_launch(void* const* d_in, const int* in_sizes, int n_in,
                              void* d_out, int out_size, void* d_ws, size_t ws_size,
                              hipStream_t stream) {
  (void)in_sizes; (void)n_in; (void)out_size; (void)ws_size;
  const float* xyz    = (const float*)d_in[0];
  const float* feats  = (const float*)d_in[1];
  const float* w0_0   = (const float*)d_in[2];
  const float* w0_1   = (const float*)d_in[3];
  const float* w0_2   = (const float*)d_in[4];
  const float* w1_0   = (const float*)d_in[5];
  const float* w1_1   = (const float*)d_in[6];
  const float* w1_2   = (const float*)d_in[7];
  const float* agg_w  = (const float*)d_in[8];
  const float* agg_b  = (const float*)d_in[9];
  const float* conf_w = (const float*)d_in[10];
  const float* cls_w  = (const float*)d_in[11];
  const float* cls_b  = (const float*)d_in[12];

  float* out_newxyz = (float*)d_out;
  float* out_h      = out_newxyz + (size_t)BB * SS * 3;
  float* out_cls    = out_h + (size_t)BB * SS * 256;

  // ---- workspace layout (256B-aligned offsets) ----
  char* ws = (char*)d_ws;
  size_t off = 0;
  auto alloc = [&](size_t bytes) {
    size_t p = off; off += (bytes + 255) & ~(size_t)255; return p;
  };
  float*     dist   = (float*)(ws + alloc((size_t)BB * NN * 4));
  float*     stats  = (float*)(ws + alloc(512 * 4));
  _Float16*  w00    = (_Float16*)(ws + alloc(96 * 64 * 2));
  _Float16*  w01    = (_Float16*)(ws + alloc(64 * 64 * 2));
  _Float16*  w02    = (_Float16*)(ws + alloc(64 * 128 * 2));
  _Float16*  w10    = (_Float16*)(ws + alloc(96 * 64 * 2));
  _Float16*  w11    = (_Float16*)(ws + alloc(64 * 64 * 2));
  _Float16*  w12    = (_Float16*)(ws + alloc(64 * 128 * 2));
  _Float16*  agg16  = (_Float16*)(ws + alloc(256 * 256 * 2));
  _Float16*  conf16 = (_Float16*)(ws + alloc(256 * 256 * 2));
  const size_t maxRows = (size_t)BB * SS * NS1;               // 524288
  _Float16*  act    = (_Float16*)(ws + alloc(maxRows * 128 * 2));
  _Float16*  Zb     = (_Float16*)(ws + alloc(maxRows * 128 * 2));
  _Float16*  concat = (_Float16*)(ws + alloc((size_t)BB * SS * 256 * 2));
  _Float16*  h16    = (_Float16*)(ws + alloc((size_t)BB * SS * 256 * 2));
  float*     cbuf   = (float*)(ws + alloc((size_t)BB * SS * 256 * 4));

  // ---- stage 1: FPS (writes new_xyz output) ----
  fps_kernel<<<BB, 256, 0, stream>>>(xyz, dist, out_newxyz);

  // ---- stage 2: weights -> f16 B-fragment layout ----
  auto conv = [&](const float* w, _Float16* dst, int K, int Kpad, int Ncols) {
    int total = Kpad * Ncols;
    convert_w_kernel<<<(total + 255) / 256, 256, 0, stream>>>(w, dst, K, Kpad, Ncols);
  };
  conv(w0_0, w00, 67, 96, 64);   conv(w0_1, w01, 64, 64, 64);
  conv(w0_2, w02, 64, 64, 128);  conv(w1_0, w10, 67, 96, 64);
  conv(w1_1, w11, 64, 64, 64);   conv(w1_2, w12, 64, 64, 128);
  conv(agg_w, agg16, 256, 256, 256);
  conv(conf_w, conf16, 256, 256, 256);

  auto gemm = [&](const _Float16* A, int lda, const _Float16* Wm, _Float16* O, int ldo,
                  const float* bias, int rows, int K, int Ncols) {
    long long tiles = (long long)(rows / 16) * (Ncols / 64);
    int blocks = (int)((tiles * 32 + 255) / 256);
    gemm_wmma_kernel<<<blocks, 256, 0, stream>>>(A, lda, Wm, O, ldo, bias, rows, K, Ncols);
  };
  // GEMM -> global BN stats -> normalize+ReLU (f16 for next layer, opt f32 out)
  auto layer = [&](const _Float16* A, int lda, const _Float16* Wm, int K, int Ncols,
                   const float* bias, int rows,
                   _Float16* o16, int ld16, float* o32, int ld32) {
    hipMemsetAsync(stats, 0, 512 * 4, stream);
    gemm(A, lda, Wm, Zb, Ncols, bias, rows, K, Ncols);
    stats_kernel<<<256, 256, 0, stream>>>(Zb, Ncols, rows, Ncols, stats);
    long long total = (long long)rows * Ncols;
    int blocks = (int)((total + 255) / 256);
    if (blocks > 16384) blocks = 16384;
    bn_relu_kernel<<<blocks, 256, 0, stream>>>(Zb, Ncols, o16, ld16, o32, ld32,
                                               stats, (size_t)rows, Ncols,
                                               1.0f / (float)rows);
  };

  // ---- stage 3: two MSG scales, sequential, shared act/Z buffers ----
  for (int sc = 0; sc < 2; ++sc) {
    const int ns = sc ? NS1 : NS0;
    const float r = sc ? RAD1 : RAD0;
    const _Float16* l0 = sc ? w10 : w00;
    const _Float16* l1 = sc ? w11 : w01;
    const _Float16* l2 = sc ? w12 : w02;
    ballq_gather_kernel<<<BB * SS / 8, 256, 0, stream>>>(xyz, feats, out_newxyz,
                                                         act, ns, r * r);
    const int rows = BB * SS * ns;
    layer(act, 96, l0, 96, 64,  nullptr, rows, act, 64,  nullptr, 0);
    layer(act, 64, l1, 64, 64,  nullptr, rows, act, 64,  nullptr, 0);
    layer(act, 64, l2, 64, 128, nullptr, rows, act, 128, nullptr, 0);
    maxpool_kernel<<<(BB * SS * 128 + 255) / 256, 256, 0, stream>>>(
        act, ns, concat, sc ? 128 : 0);
  }

  // ---- stage 4: aggregation (writes f32 h output) + confidence + cls ----
  layer(concat, 256, agg16, 256, 256, agg_b, BB * SS, h16, 256, out_h, 256);
  layer(h16, 256, conf16, 256, 256, nullptr, BB * SS, nullptr, 0, cbuf, 256);
  cls_kernel<<<(BB * SS * 3 + 255) / 256, 256, 0, stream>>>(cbuf, cls_w, cls_b, out_cls);
}
